// Qwen3NextSparseMoeBlock_54357106098375
// MI455X (gfx1250) — matrix operand order repaired and checked by reading, not compiled, not run on hardware
//
#include <hip/hip_runtime.h>

// Problem dims (fixed by reference)
#define T_TOK 2048   // B*S
#define HDIM  2048
#define NEXP  16
#define FDIM  1024
#define SHF   5632
#define TOPK  4

// GEMM tiling: 256 threads = 8 waves; block tile 128x64; wave tile 32x32 (2x2 WMMA); K step 32
#define TM 128
#define TN 64
#define TK 32

#if __has_builtin(__builtin_amdgcn_global_load_async_to_lds_b128)
#define HAVE_ASYNC_LDS 1
#else
#define HAVE_ASYNC_LDS 0
#endif

typedef __attribute__((ext_vector_type(4)))  int    v4i;
typedef __attribute__((ext_vector_type(4)))  float  v4f;
typedef __attribute__((ext_vector_type(4)))  __bf16 v4bf;
typedef __attribute__((ext_vector_type(8)))  __bf16 v8bf;
typedef __attribute__((ext_vector_type(16))) __bf16 v16bf;
typedef __attribute__((ext_vector_type(8)))  float  v8f;

__device__ __forceinline__ v4bf cvt4(float4 a) {
  v4f f; f[0] = a.x; f[1] = a.y; f[2] = a.z; f[3] = a.w;
  return __builtin_convertvector(f, v4bf);   // native v_cvt_pk_bf16_f32
}

__device__ __forceinline__ float silu_f(float g) {
  return g / (1.0f + __expf(-g));
}

__device__ __forceinline__ unsigned short f32_to_bf16(float f) {
  v4f x; x[0] = f; x[1] = 0.f; x[2] = 0.f; x[3] = 0.f;
  v4bf h = __builtin_convertvector(x, v4bf);
  return __builtin_bit_cast(unsigned short, h[0]);
}

__device__ __forceinline__ v8f wmma_bf16(v16bf a, v16bf b, v8f c) {
  return __builtin_amdgcn_wmma_f32_16x16x32_bf16(
      /*neg_a=*/false, a, /*neg_b=*/false, b,
      /*c_mod=*/(short)0, c, /*reuse_a=*/false, /*reuse_b=*/false);
}

__device__ __forceinline__ void wait_async0() {
#if __has_builtin(__builtin_amdgcn_s_wait_asynccnt)
  __builtin_amdgcn_s_wait_asynccnt(0);
#else
  asm volatile("s_wait_asynccnt 0x0" ::: "memory");
#endif
}

// ---- register staging + LDS commit (double-buffered pipeline) ----------------

struct StageA   { float4  v[4]; };   // 128x32 fp32 tile, 16 floats/thread
struct StageAbf { ushort4 v[4]; };   // 128x32 bf16 tile
struct StageB   { float4  lo, hi; }; // 32x64 fp32 tile, 8 floats/thread

__device__ __forceinline__ StageA stage_a_f32(const float* __restrict__ A, int lda,
                                              int row0, int k0, int tid) {
  StageA s;
#pragma unroll
  for (int i = 0; i < 4; ++i) {          // issue all loads before any use
    int id = tid + 256 * i, row = id >> 3, col = (id & 7) << 2;
    s.v[i] = *reinterpret_cast<const float4*>(A + (size_t)(row0 + row) * lda + k0 + col);
  }
  return s;
}

__device__ __forceinline__ void commit_a_f32(const StageA& s, unsigned short* As, int tid) {
#pragma unroll
  for (int i = 0; i < 4; ++i) {
    int id = tid + 256 * i, row = id >> 3, col = (id & 7) << 2;
    *reinterpret_cast<v4bf*>(&As[row * TK + col]) = cvt4(s.v[i]);
  }
}

__device__ __forceinline__ StageAbf stage_a_bf16(const unsigned short* __restrict__ A,
                                                 int lda, int row0, int k0, int tid) {
  StageAbf s;
#pragma unroll
  for (int i = 0; i < 4; ++i) {
    int id = tid + 256 * i, row = id >> 3, col = (id & 7) << 2;
    s.v[i] = *reinterpret_cast<const ushort4*>(A + (size_t)(row0 + row) * lda + k0 + col);
  }
  return s;
}

__device__ __forceinline__ void commit_a_bf16(const StageAbf& s, unsigned short* As, int tid) {
#pragma unroll
  for (int i = 0; i < 4; ++i) {
    int id = tid + 256 * i, row = id >> 3, col = (id & 7) << 2;
    *reinterpret_cast<ushort4*>(&As[row * TK + col]) = s.v[i];
  }
}

#if HAVE_ASYNC_LDS
// bf16 A tile: DMA 128x32 (8 KB) straight into LDS, no VGPR round-trip.
// 512 16-byte chunks / 256 threads = 2 async b128 per thread; tracked by ASYNCcnt.
__device__ __forceinline__ void async_load_a_bf16(const unsigned short* __restrict__ A,
                                                  int lda, int row0, int k0,
                                                  unsigned short* As, int tid) {
#pragma unroll
  for (int i = 0; i < 2; ++i) {
    int id  = tid + 256 * i;           // 0..511
    int row = id >> 2;                 // 4 chunks per 32-elem row
    int col = (id & 3) << 3;           // 8 ushorts = 16 B
    const unsigned short* g = A + (size_t)(row0 + row) * lda + k0 + col;
    __builtin_amdgcn_global_load_async_to_lds_b128(
        (__attribute__((address_space(1))) v4i*)(g),
        (__attribute__((address_space(3))) v4i*)(&As[row * TK + col]),
        /*offset=*/0, /*cpol=*/0);
  }
}
#endif

__device__ __forceinline__ StageB stage_b_f32(const float* __restrict__ B, int ldb,
                                              int k0, int col0, int tid) {
  int k = tid >> 3, n = (tid & 7) << 3;
  const float* src = B + (size_t)(k0 + k) * ldb + col0 + n;
  StageB s;
  s.lo = *reinterpret_cast<const float4*>(src);
  s.hi = *reinterpret_cast<const float4*>(src + 4);
  return s;
}

// LDS stores K-transposed: Bt[n][k], stride TK
__device__ __forceinline__ void commit_b_f32(const StageB& s, unsigned short* Bt, int tid) {
  int k = tid >> 3, n = (tid & 7) << 3;
  v4bf h0 = cvt4(s.lo), h1 = cvt4(s.hi);
#pragma unroll
  for (int i = 0; i < 4; ++i)
    Bt[(n + i) * TK + k] = __builtin_bit_cast(unsigned short, h0[i]);
#pragma unroll
  for (int i = 0; i < 4; ++i)
    Bt[(n + 4 + i) * TK + k] = __builtin_bit_cast(unsigned short, h1[i]);
}

// ---- WMMA fragment loaders (documented 16-bit layouts) -----------------------

// A 16x32 (MxK): lane<16 row=lane, K 0..7 & 16..23; lane>=16: K 8..15 & 24..31
__device__ __forceinline__ v16bf frag_a(const unsigned short* As, int lane, int rbase) {
  int row = rbase + (lane & 15);
  int kh  = (lane >> 4) << 3;           // 0 or 8
  const __bf16* p = reinterpret_cast<const __bf16*>(As + row * TK);
  v8bf lo = *reinterpret_cast<const v8bf*>(p + kh);
  v8bf hi = *reinterpret_cast<const v8bf*>(p + kh + 16);
  v16bf r;
#pragma unroll
  for (int i = 0; i < 8; ++i) { r[i] = lo[i]; r[i + 8] = hi[i]; }
  return r;
}

// B 32x16 (KxN): lane<16 col=lane, K 0..7 & 16..23; lane>=16: K 8..15 & 24..31
__device__ __forceinline__ v16bf frag_b(const unsigned short* Bt, int lane, int cbase) {
  int col = cbase + (lane & 15);
  int kh  = (lane >> 4) << 3;           // 0 or 8
  const __bf16* p = reinterpret_cast<const __bf16*>(Bt + col * TK + kh);
  v8bf lo = *reinterpret_cast<const v8bf*>(p);
  v8bf hi = *reinterpret_cast<const v8bf*>(p + 16);
  v16bf r;
#pragma unroll
  for (int i = 0; i < 8; ++i) { r[i] = lo[i]; r[i + 8] = hi[i]; }
  return r;
}

// ---- Kernel 1: router (softmax -> top4 -> renorm) + shared-expert gate -------

__global__ __launch_bounds__(256) void router_kernel(
    const float* __restrict__ x, const float* __restrict__ w_router,
    const float* __restrict__ w_sh_eg,
    float* __restrict__ combine, float* __restrict__ gatev) {
  __shared__ float red[256];
  __shared__ float red2[256];
  __shared__ float logit_s[NEXP];
  const int t = blockIdx.x;
  const int tid = threadIdx.x;
  const int e = tid & 15, chunk = tid >> 4;   // 16 experts x 16 chunks
  const float* xr = x + (size_t)t * HDIM;

  float part = 0.f, part2 = 0.f;
  const int span = HDIM / 16;
  for (int h = chunk * span; h < (chunk + 1) * span; ++h)
    part += xr[h] * w_router[h * NEXP + e];
  for (int h = tid; h < HDIM; h += 256)
    part2 += xr[h] * w_sh_eg[h];
  red[tid] = part;
  red2[tid] = part2;
  __syncthreads();

  for (int s = 128; s > 0; s >>= 1) {
    if (tid < s) red2[tid] += red2[tid + s];
    __syncthreads();
  }
  if (tid < NEXP) {
    float l = 0.f;
    for (int c = 0; c < 16; ++c) l += red[c * 16 + tid];
    logit_s[tid] = l;
  }
  __syncthreads();

  if (tid == 0) {
    float p[NEXP], m = -1e30f, sum = 0.f;
    for (int i = 0; i < NEXP; ++i) m = fmaxf(m, logit_s[i]);
    for (int i = 0; i < NEXP; ++i) { p[i] = __expf(logit_s[i] - m); sum += p[i]; }
    for (int i = 0; i < NEXP; ++i) p[i] /= sum;
    float w[NEXP]; bool tk[NEXP]; float csum = 0.f;
    for (int i = 0; i < NEXP; ++i) { w[i] = 0.f; tk[i] = false; }
    for (int k = 0; k < TOPK; ++k) {
      int best = -1; float bv = -1.f;
      for (int i = 0; i < NEXP; ++i)
        if (!tk[i] && p[i] > bv) { bv = p[i]; best = i; }
      tk[best] = true; w[best] = bv; csum += bv;
    }
    for (int i = 0; i < NEXP; ++i) combine[(size_t)t * NEXP + i] = w[i] / csum;
    gatev[t] = 1.f / (1.f + __expf(-red2[0]));
  }
}

// ---- Kernel 2: dual GEMM (gate & up share A) + SwiGLU epilogue --------------
// out = silu(A@Wg) * (A@Wu) [* combine[:,e]]  -> bf16 [T, N];  K = HDIM.

template <bool USE_COMBINE>
__global__ __launch_bounds__(256) void gateup_kernel(
    const float* __restrict__ A,        // [T, HDIM] fp32
    const float* __restrict__ Wg,       // [HDIM, N] fp32 (already expert-offset)
    const float* __restrict__ Wu,       // [HDIM, N] fp32
    const float* __restrict__ combine,  // [T, NEXP] (moe only)
    unsigned short* __restrict__ out,   // [T, N] bf16
    int N, int e) {
  __shared__ __align__(16) unsigned short As[2][TM * TK];
  __shared__ __align__(16) unsigned short Bg[2][TN * TK];
  __shared__ __align__(16) unsigned short Bu[2][TN * TK];
  const int tid = threadIdx.x, lane = tid & 31, wave = tid >> 5;
  const int wr = (wave >> 1) * 32, wc = (wave & 1) * 32;
  const int row0 = blockIdx.x * TM, col0 = blockIdx.y * TN;

  v8f accg[2][2] = {}; v8f accu[2][2] = {};

  {
    StageA sa = stage_a_f32(A, HDIM, row0, 0, tid);
    StageB sg = stage_b_f32(Wg, N, 0, col0, tid);
    StageB su = stage_b_f32(Wu, N, 0, col0, tid);
    commit_a_f32(sa, As[0], tid);
    commit_b_f32(sg, Bg[0], tid);
    commit_b_f32(su, Bu[0], tid);
  }
  __syncthreads();

  int buf = 0;
  for (int k0 = 0; k0 < HDIM; k0 += TK) {
    const bool has_next = (k0 + TK) < HDIM;
    StageA na; StageB ng, nu;
    if (has_next) {                       // issue next tile's global loads first
      na = stage_a_f32(A, HDIM, row0, k0 + TK, tid);
      ng = stage_b_f32(Wg, N, k0 + TK, col0, tid);
      nu = stage_b_f32(Wu, N, k0 + TK, col0, tid);
    }
    // compute current buffer
    v16bf a0 = frag_a(As[buf], lane, wr);
    v16bf a1 = frag_a(As[buf], lane, wr + 16);
    v16bf g0 = frag_b(Bg[buf], lane, wc);
    v16bf g1 = frag_b(Bg[buf], lane, wc + 16);
    v16bf u0 = frag_b(Bu[buf], lane, wc);
    v16bf u1 = frag_b(Bu[buf], lane, wc + 16);
    accg[0][0] = wmma_bf16(a0, g0, accg[0][0]);
    accg[0][1] = wmma_bf16(a0, g1, accg[0][1]);
    accg[1][0] = wmma_bf16(a1, g0, accg[1][0]);
    accg[1][1] = wmma_bf16(a1, g1, accg[1][1]);
    accu[0][0] = wmma_bf16(a0, u0, accu[0][0]);
    accu[0][1] = wmma_bf16(a0, u1, accu[0][1]);
    accu[1][0] = wmma_bf16(a1, u0, accu[1][0]);
    accu[1][1] = wmma_bf16(a1, u1, accu[1][1]);
    if (has_next) {                       // commit into the other buffer
      commit_a_f32(na, As[buf ^ 1], tid);
      commit_b_f32(ng, Bg[buf ^ 1], tid);
      commit_b_f32(nu, Bu[buf ^ 1], tid);
      __syncthreads();                    // single barrier per K-step
      buf ^= 1;
    }
  }

  const int rowoff = (lane >> 4) << 3;    // C layout: lanes 16-31 hold M=r+8
  const int coloff = lane & 15;
#pragma unroll
  for (int i = 0; i < 2; ++i)
#pragma unroll
    for (int j = 0; j < 2; ++j)
#pragma unroll
      for (int r = 0; r < 8; ++r) {
        int row = row0 + wr + i * 16 + rowoff + r;
        int col = col0 + wc + j * 16 + coloff;
        float v = silu_f(accg[i][j][r]) * accu[i][j][r];
        if (USE_COMBINE) v *= combine[(size_t)row * NEXP + e];
        out[(size_t)row * N + col] = f32_to_bf16(v);
      }
}

// ---- Kernel 3: down projection -----------------------------------------------
// FINAL=false: moe_out[T,H] += act @ B   (per-expert accumulate, launches serialize)
// FINAL=true : out[T,H] = moe_out + gate[t] * (sg @ B)
// bf16 A tile goes through the ASYNCcnt global->LDS DMA path when available.

template <bool FINAL>
__global__ __launch_bounds__(256) void down_kernel(
    const unsigned short* __restrict__ A,  // [T, K] bf16
    const float* __restrict__ B,           // [K, HDIM] fp32
    float* __restrict__ moe_out,           // [T, HDIM]
    const float* __restrict__ gatev,       // [T]
    float* __restrict__ out,               // [T, HDIM]
    int K) {
  __shared__ __align__(16) unsigned short As[2][TM * TK];
  __shared__ __align__(16) unsigned short Bt[2][TN * TK];
  const int tid = threadIdx.x, lane = tid & 31, wave = tid >> 5;
  const int wr = (wave >> 1) * 32, wc = (wave & 1) * 32;
  const int row0 = blockIdx.x * TM, col0 = blockIdx.y * TN;

  v8f acc[2][2] = {};

#if HAVE_ASYNC_LDS
  async_load_a_bf16(A, K, row0, 0, As[0], tid);
  {
    StageB sb = stage_b_f32(B, HDIM, 0, col0, tid);
    commit_b_f32(sb, Bt[0], tid);
  }
  wait_async0();
  __syncthreads();

  int buf = 0;
  for (int k0 = 0; k0 < K; k0 += TK) {
    const bool has_next = (k0 + TK) < K;
    StageB nb;
    if (has_next) {
      // DMA next A tile straight into the other LDS buffer while we compute
      async_load_a_bf16(A, K, row0, k0 + TK, As[buf ^ 1], tid);
      nb = stage_b_f32(B, HDIM, k0 + TK, col0, tid);
    }
    v16bf a0 = frag_a(As[buf], lane, wr);
    v16bf a1 = frag_a(As[buf], lane, wr + 16);
    v16bf b0 = frag_b(Bt[buf], lane, wc);
    v16bf b1 = frag_b(Bt[buf], lane, wc + 16);
    acc[0][0] = wmma_bf16(a0, b0, acc[0][0]);
    acc[0][1] = wmma_bf16(a0, b1, acc[0][1]);
    acc[1][0] = wmma_bf16(a1, b0, acc[1][0]);
    acc[1][1] = wmma_bf16(a1, b1, acc[1][1]);
    if (has_next) {
      commit_b_f32(nb, Bt[buf ^ 1], tid);
      wait_async0();
      __syncthreads();
      buf ^= 1;
    }
  }
#else
  {
    StageAbf sa = stage_a_bf16(A, K, row0, 0, tid);
    StageB   sb = stage_b_f32(B, HDIM, 0, col0, tid);
    commit_a_bf16(sa, As[0], tid);
    commit_b_f32(sb, Bt[0], tid);
  }
  __syncthreads();

  int buf = 0;
  for (int k0 = 0; k0 < K; k0 += TK) {
    const bool has_next = (k0 + TK) < K;
    StageAbf na; StageB nb;
    if (has_next) {
      na = stage_a_bf16(A, K, row0, k0 + TK, tid);
      nb = stage_b_f32(B, HDIM, k0 + TK, col0, tid);
    }
    v16bf a0 = frag_a(As[buf], lane, wr);
    v16bf a1 = frag_a(As[buf], lane, wr + 16);
    v16bf b0 = frag_b(Bt[buf], lane, wc);
    v16bf b1 = frag_b(Bt[buf], lane, wc + 16);
    acc[0][0] = wmma_bf16(a0, b0, acc[0][0]);
    acc[0][1] = wmma_bf16(a0, b1, acc[0][1]);
    acc[1][0] = wmma_bf16(a1, b0, acc[1][0]);
    acc[1][1] = wmma_bf16(a1, b1, acc[1][1]);
    if (has_next) {
      commit_a_bf16(na, As[buf ^ 1], tid);
      commit_b_f32(nb, Bt[buf ^ 1], tid);
      __syncthreads();
      buf ^= 1;
    }
  }
#endif

  const int rowoff = (lane >> 4) << 3;
  const int coloff = lane & 15;
#pragma unroll
  for (int i = 0; i < 2; ++i)
#pragma unroll
    for (int j = 0; j < 2; ++j)
#pragma unroll
      for (int r = 0; r < 8; ++r) {
        int row = row0 + wr + i * 16 + rowoff + r;
        int col = col0 + wc + j * 16 + coloff;
        size_t idx = (size_t)row * HDIM + col;
        if (FINAL)
          out[idx] = moe_out[idx] + gatev[row] * acc[i][j][r];
        else
          moe_out[idx] += acc[i][j][r];
      }
}

// ---- host launch -------------------------------------------------------------

extern "C" void kernel_launch(void* const* d_in, const int* in_sizes, int n_in,
                              void* d_out, int out_size, void* d_ws, size_t ws_size,
                              hipStream_t stream) {
  (void)in_sizes; (void)n_in; (void)out_size; (void)ws_size;
  const float* x         = (const float*)d_in[0];  // [T, H]
  const float* w_router  = (const float*)d_in[1];  // [H, E]
  const float* w_gate    = (const float*)d_in[2];  // [E, H, F]
  const float* w_up      = (const float*)d_in[3];  // [E, H, F]
  const float* w_down    = (const float*)d_in[4];  // [E, F, H]
  const float* w_sh_gate = (const float*)d_in[5];  // [H, SHF]
  const float* w_sh_up   = (const float*)d_in[6];  // [H, SHF]
  const float* w_sh_down = (const float*)d_in[7];  // [SHF, H]
  const float* w_sh_eg   = (const float*)d_in[8];  // [H, 1]
  float* out = (float*)d_out;                      // [T, H]

  // workspace layout (~43 MB)
  char* p = (char*)d_ws;
  float* combine = (float*)p;           p += (size_t)T_TOK * NEXP * sizeof(float);
  float* gatev   = (float*)p;           p += (size_t)T_TOK * sizeof(float);
  float* moe_out = (float*)p;           p += (size_t)T_TOK * HDIM * sizeof(float);
  unsigned short* act = (unsigned short*)p; p += (size_t)T_TOK * FDIM * sizeof(unsigned short);
  unsigned short* sg  = (unsigned short*)p; // [T, SHF] bf16

  (void)hipMemsetAsync(moe_out, 0, (size_t)T_TOK * HDIM * sizeof(float), stream);

  // 1) router + shared-expert sigmoid gate
  router_kernel<<<T_TOK, 256, 0, stream>>>(x, w_router, w_sh_eg, combine, gatev);

  // 2) shared expert gate/up -> sg (bf16)
  gateup_kernel<false><<<dim3(T_TOK / TM, SHF / TN), 256, 0, stream>>>(
      x, w_sh_gate, w_sh_up, nullptr, sg, SHF, 0);

  // 3) experts: (gate+up with SwiGLU*combine) -> act, then down accumulate
  for (int e = 0; e < NEXP; ++e) {
    const float* Wg = w_gate + (size_t)e * HDIM * FDIM;
    const float* Wu = w_up   + (size_t)e * HDIM * FDIM;
    const float* Wd = w_down + (size_t)e * FDIM * HDIM;
    gateup_kernel<true><<<dim3(T_TOK / TM, FDIM / TN), 256, 0, stream>>>(
        x, Wg, Wu, combine, act, FDIM, e);
    down_kernel<false><<<dim3(T_TOK / TM, HDIM / TN), 256, 0, stream>>>(
        act, Wd, moe_out, nullptr, nullptr, FDIM);
  }

  // 4) shared down + final combine: out = moe_out + gate[t] * (sg @ w_sh_down)
  down_kernel<true><<<dim3(T_TOK / TM, HDIM / TN), 256, 0, stream>>>(
      sg, w_sh_down, moe_out, gatev, out, SHF);
}
